// AttentionBase_11424613008041
// MI455X (gfx1250) — matrix operand order repaired and checked by reading, not compiled, not run on hardware
//
#include <hip/hip_runtime.h>
#include <math.h>

// ---------------------------------------------------------------------------
// Shapes (fixed by the reference): B=4, N=2048, H=16, d=64, D=1024
// ---------------------------------------------------------------------------
#define BATCH   4
#define SEQ     2048
#define NHEADS  16
#define HDIM    64
#define DMODEL  1024
#define LN_EPS  1e-5f

typedef __attribute__((ext_vector_type(16))) __bf16 bf16x16;
typedef __attribute__((ext_vector_type(8)))  float  f32x8;

union Frag {
    bf16x16  v;
    unsigned u[8];
    uint4    q2[2];   // two 16B groups -> ds_load_b128 / global_load_b128
};

// Native f32 -> bf16 converts (RNE fptrunc on gfx1250).
__device__ __forceinline__ unsigned pack2(float a, float b) {
    union { unsigned u; __bf16 h[2]; } p;
    p.h[0] = (__bf16)a;
    p.h[1] = (__bf16)b;
    return p.u;
}
__device__ __forceinline__ unsigned short bf16b(float f) {
    union { unsigned short s; __bf16 h; } p;
    p.h = (__bf16)f;
    return p.s;
}

__device__ __forceinline__ f32x8 wmma_bf16(const Frag& a, const Frag& b, f32x8 c) {
    return __builtin_amdgcn_wmma_f32_16x16x32_bf16(
        /*neg_a=*/false, a.v, /*neg_b=*/false, b.v,
        /*c_mod=*/(short)0, c, /*reuse_a=*/false, /*reuse_b=*/false);
}

// --- DPP16 ROW_XMASK cross-lane (16-lane row butterfly, no LDS traffic) ---
template <int XMASK>
__device__ __forceinline__ float dpp_xor(float x) {
    const int xi = __builtin_bit_cast(int, x);
    const int r  = __builtin_amdgcn_update_dpp(0, xi, 0x160 | XMASK,
                                               /*row_mask=*/0xF, /*bank_mask=*/0xF,
                                               /*bound_ctrl=*/true);
    return __builtin_bit_cast(float, r);
}
__device__ __forceinline__ float row16_max(float x) {
    x = fmaxf(x, dpp_xor<1>(x));
    x = fmaxf(x, dpp_xor<2>(x));
    x = fmaxf(x, dpp_xor<4>(x));
    x = fmaxf(x, dpp_xor<8>(x));
    return x;
}
__device__ __forceinline__ float row16_sum(float x) {
    x += dpp_xor<1>(x);
    x += dpp_xor<2>(x);
    x += dpp_xor<4>(x);
    x += dpp_xor<8>(x);
    return x;
}

// ---------------------------------------------------------------------------
// Kernel 1: flash attention. Grid: 2048 blocks (b*h*32 q-tiles), 128 threads.
// LDS: Ks 64x64 bf16 [key][d] (8KB) | Vts 64x64 bf16 [d][key] (8KB)
//    | Ps 64x64 bf16 [qrow][key]  (8KB)
// ---------------------------------------------------------------------------
__global__ __launch_bounds__(128) void attn_kernel(
    const float* __restrict__ q, const float* __restrict__ k,
    const float* __restrict__ v, float* __restrict__ o)
{
    extern __shared__ char smem[];
    unsigned*       Ks  = (unsigned*)smem;               // uint idx = key*32 + d/2
    unsigned*       Vts = (unsigned*)(smem + 8192);      // uint idx = d*32 + key/2
    unsigned short* Vth = (unsigned short*)(smem + 8192);
    unsigned short* Ps  = (unsigned short*)(smem + 16384);
    unsigned*       Psu = (unsigned*)(smem + 16384);

    const int wg   = blockIdx.x;
    const int mt   = wg & 31;         // query tile (2048/64)
    const int bh   = wg >> 5;
    const int h    = bh & (NHEADS - 1);
    const int b    = bh >> 4;

    const int tid  = threadIdx.x;
    const int wave = tid >> 5;
    const int lane = tid & 31;
    const int lm   = lane & 15;
    const int half = lane >> 4;

    const size_t hoff = (size_t)h * HDIM;
    const float* qb = q + (size_t)b * SEQ * DMODEL + hoff;
    const float* kb = k + (size_t)b * SEQ * DMODEL + hoff;
    const float* vb = v + (size_t)b * SEQ * DMODEL + hoff;

    // --- Q A-fragments (rows mt*64 + wave*16 + lm), scaled by d^-0.5 ---
    const int   qrow  = mt * 64 + wave * 16 + lm;
    const float scale = 0.125f;  // 1/sqrt(64)
    Frag qa[2];
#pragma unroll
    for (int f = 0; f < 2; ++f) {
        const float* p0 = qb + (size_t)qrow * DMODEL + f * 32 + half * 8;
        const float4 a0 = *(const float4*)(p0);
        const float4 a1 = *(const float4*)(p0 + 4);
        const float4 a2 = *(const float4*)(p0 + 16);
        const float4 a3 = *(const float4*)(p0 + 20);
        qa[f].u[0] = pack2(a0.x * scale, a0.y * scale);
        qa[f].u[1] = pack2(a0.z * scale, a0.w * scale);
        qa[f].u[2] = pack2(a1.x * scale, a1.y * scale);
        qa[f].u[3] = pack2(a1.z * scale, a1.w * scale);
        qa[f].u[4] = pack2(a2.x * scale, a2.y * scale);
        qa[f].u[5] = pack2(a2.z * scale, a2.w * scale);
        qa[f].u[6] = pack2(a3.x * scale, a3.y * scale);
        qa[f].u[7] = pack2(a3.z * scale, a3.w * scale);
    }

    f32x8 oacc[4] = {};
    float mrow[8], lrow[8];
#pragma unroll
    for (int r = 0; r < 8; ++r) { mrow[r] = -INFINITY; lrow[r] = 0.0f; }

    for (int jt = 0; jt < SEQ / 64; ++jt) {
        __syncthreads();   // protect Ks/Vts/Ps from previous iteration readers
        // ---- stage K (row-major bf16) and V (transposed bf16) into LDS ----
#pragma unroll
        for (int i = 0; i < 8; ++i) {
            const int idx4 = tid + i * 128;      // float4 index, 1024 total
            const int row  = idx4 >> 4;          // key within tile
            const int c4   = idx4 & 15;          // float4 column group
            const float4 kk4 = *(const float4*)(kb + (size_t)(jt * 64 + row) * DMODEL + c4 * 4);
            Ks[row * 32 + c4 * 2    ] = pack2(kk4.x, kk4.y);
            Ks[row * 32 + c4 * 2 + 1] = pack2(kk4.z, kk4.w);
            const float4 vv4 = *(const float4*)(vb + (size_t)(jt * 64 + row) * DMODEL + c4 * 4);
            Vth[(c4 * 4    ) * 64 + row] = bf16b(vv4.x);
            Vth[(c4 * 4 + 1) * 64 + row] = bf16b(vv4.y);
            Vth[(c4 * 4 + 2) * 64 + row] = bf16b(vv4.z);
            Vth[(c4 * 4 + 3) * 64 + row] = bf16b(vv4.w);
        }
        // prefetch next tile (one cacheline probe per lane) while we compute
        if (jt + 1 < SEQ / 64) {
            const size_t nroff = (size_t)(jt * 64 + 64 + (tid >> 1)) * DMODEL + (tid & 1) * 32;
            __builtin_prefetch(kb + nroff, 0, 3);
            __builtin_prefetch(vb + nroff, 0, 3);
        }
        __syncthreads();

        // ---- S = Q * K^T : 4 tiles of 16x16 per wave ----
        f32x8 s[4] = {};
#pragma unroll
        for (int f = 0; f < 2; ++f) {
            Frag bk[4];
#pragma unroll
            for (int t = 0; t < 4; ++t) {
                const unsigned* p = Ks + (t * 16 + lm) * 32 + f * 16 + half * 4;
                bk[t].q2[0] = *(const uint4*)(p);
                bk[t].q2[1] = *(const uint4*)(p + 8);
            }
#pragma unroll
            for (int t = 0; t < 4; ++t)
                s[t] = wmma_bf16(qa[f], bk[t], s[t]);
        }

        // ---- online softmax (rows r + 8*half; DPP16 row butterflies) ----
        float mnew[8], alpha[8];
#pragma unroll
        for (int r = 0; r < 8; ++r) {
            float mx = fmaxf(fmaxf(s[0][r], s[1][r]), fmaxf(s[2][r], s[3][r]));
            mx       = row16_max(mx);
            mnew[r]  = fmaxf(mrow[r], mx);
            alpha[r] = __expf(mrow[r] - mnew[r]);
            mrow[r]  = mnew[r];
        }
#pragma unroll
        for (int r = 0; r < 8; ++r) {
            float rs = 0.0f;
#pragma unroll
            for (int t = 0; t < 4; ++t) {
                s[t][r] = __expf(s[t][r] - mnew[r]);
                rs += s[t][r];
            }
            rs = row16_sum(rs);
            lrow[r] = lrow[r] * alpha[r] + rs;
#pragma unroll
            for (int t = 0; t < 4; ++t) oacc[t][r] *= alpha[r];
        }

        // ---- P: C-layout -> LDS (bf16) -> A-layout ----
#pragma unroll
        for (int t = 0; t < 4; ++t)
#pragma unroll
            for (int r = 0; r < 8; ++r)
                Ps[(wave * 16 + r + half * 8) * 64 + t * 16 + lm] = bf16b(s[t][r]);
        __syncthreads();

        // ---- O += P * V ----
#pragma unroll
        for (int f = 0; f < 2; ++f) {
            Frag pa;
            {
                const unsigned* p = Psu + (wave * 16 + lm) * 32 + f * 16 + half * 4;
                pa.q2[0] = *(const uint4*)(p);       // kk groups 0..3
                pa.q2[1] = *(const uint4*)(p + 8);   // kk groups 4..7
            }
            Frag bv[4];
#pragma unroll
            for (int t = 0; t < 4; ++t) {
                const unsigned* p = Vts + (t * 16 + lm) * 32 + f * 16 + half * 4;
                bv[t].q2[0] = *(const uint4*)(p);
                bv[t].q2[1] = *(const uint4*)(p + 8);
            }
#pragma unroll
            for (int t = 0; t < 4; ++t)
                oacc[t] = wmma_bf16(pa, bv[t], oacc[t]);
        }
    }

    // ---- epilogue: O /= l, store f32 to [B,N,D] scratch ----
#pragma unroll
    for (int r = 0; r < 8; ++r) {
        const float inv = 1.0f / lrow[r];
        const int   row = mt * 64 + wave * 16 + r + half * 8;
#pragma unroll
        for (int t = 0; t < 4; ++t)
            o[((size_t)b * SEQ + row) * DMODEL + h * 64 + t * 16 + lm] = oacc[t][r] * inv;
    }
}

// ---------------------------------------------------------------------------
// Kernel 2: Y = X * W^T.  X:[8192,1024] f32 scratch, W:[1024,1024].
// Grid (16,128) = (col tiles, row tiles) of 64x64, 128 threads (4 waves).
// LDS: Xs 64x32 bf16 (4KB) | Ws 64x32 bf16 (4KB)
// ---------------------------------------------------------------------------
__global__ __launch_bounds__(128) void proj_kernel(
    const float* __restrict__ X, const float* __restrict__ W,
    float* __restrict__ Y)
{
    extern __shared__ char smem[];
    unsigned* Xs = (unsigned*)smem;            // uint idx = row*16 + k/2
    unsigned* Ws = (unsigned*)(smem + 4096);

    const int jt   = blockIdx.x;   // 0..15  (output col tile)
    const int it   = blockIdx.y;   // 0..127 (row tile)
    const int tid  = threadIdx.x;
    const int wave = tid >> 5;
    const int lane = tid & 31;
    const int lm   = lane & 15;
    const int half = lane >> 4;

    f32x8 acc[4] = {};

    for (int kk = 0; kk < DMODEL / 32; ++kk) {
        __syncthreads();
#pragma unroll
        for (int i = 0; i < 4; ++i) {
            const int idx = tid + i * 128;     // float4 index, 512 total
            const int row = idx >> 3;          // 8 float4 per 32-wide row
            const int c4  = idx & 7;
            const float4 x4 = *(const float4*)(X + (size_t)(it * 64 + row) * DMODEL + kk * 32 + c4 * 4);
            Xs[row * 16 + c4 * 2    ] = pack2(x4.x, x4.y);
            Xs[row * 16 + c4 * 2 + 1] = pack2(x4.z, x4.w);
            const float4 w4 = *(const float4*)(W + (size_t)(jt * 64 + row) * DMODEL + kk * 32 + c4 * 4);
            Ws[row * 16 + c4 * 2    ] = pack2(w4.x, w4.y);
            Ws[row * 16 + c4 * 2 + 1] = pack2(w4.z, w4.w);
        }
        if (kk + 1 < DMODEL / 32) {
            __builtin_prefetch(X + (size_t)(it * 64 + (tid >> 1)) * DMODEL + (kk + 1) * 32 + (tid & 1) * 16, 0, 3);
            __builtin_prefetch(W + (size_t)(jt * 64 + (tid >> 1)) * DMODEL + (kk + 1) * 32 + (tid & 1) * 16, 0, 3);
        }
        __syncthreads();

        Frag a;
        {
            const unsigned* p = Xs + (wave * 16 + lm) * 16 + half * 4;
            a.q2[0] = *(const uint4*)(p);
            a.q2[1] = *(const uint4*)(p + 8);
        }
        Frag bw[4];
#pragma unroll
        for (int t = 0; t < 4; ++t) {
            const unsigned* p = Ws + (t * 16 + lm) * 16 + half * 4;
            bw[t].q2[0] = *(const uint4*)(p);
            bw[t].q2[1] = *(const uint4*)(p + 8);
        }
#pragma unroll
        for (int t = 0; t < 4; ++t)
            acc[t] = wmma_bf16(a, bw[t], acc[t]);
    }

#pragma unroll
    for (int r = 0; r < 8; ++r) {
        const int row = it * 64 + wave * 16 + r + half * 8;
#pragma unroll
        for (int t = 0; t < 4; ++t)
            Y[(size_t)row * DMODEL + jt * 64 + t * 16 + lm] = acc[t][r];
    }
}

// ---------------------------------------------------------------------------
// Kernel 3: LayerNorm (bias-free, biased variance) * g. One block per row.
// ---------------------------------------------------------------------------
__global__ __launch_bounds__(256) void ln_kernel(
    const float* __restrict__ Y, const float* __restrict__ g,
    float* __restrict__ out)
{
    __shared__ float s1[8], s2[8];
    const int row  = blockIdx.x;
    const int tid  = threadIdx.x;
    const int wave = tid >> 5;
    const int lane = tid & 31;

    const float4 x4 = *(const float4*)(Y + (size_t)row * DMODEL + tid * 4);
    float sum = x4.x + x4.y + x4.z + x4.w;
    float sq  = x4.x * x4.x + x4.y * x4.y + x4.z * x4.z + x4.w * x4.w;

#pragma unroll
    for (int off = 1; off < 32; off <<= 1) {
        sum += __shfl_xor(sum, off, 32);
        sq  += __shfl_xor(sq,  off, 32);
    }
    if (lane == 0) { s1[wave] = sum; s2[wave] = sq; }
    __syncthreads();
    float ts = 0.0f, tq = 0.0f;
#pragma unroll
    for (int w = 0; w < 8; ++w) { ts += s1[w]; tq += s2[w]; }

    const float mean = ts * (1.0f / DMODEL);
    const float var  = tq * (1.0f / DMODEL) - mean * mean;
    const float inv  = rsqrtf(var + LN_EPS);

    const float4 g4 = *(const float4*)(g + tid * 4);
    float4 o4;
    o4.x = (x4.x - mean) * inv * g4.x;
    o4.y = (x4.y - mean) * inv * g4.y;
    o4.z = (x4.z - mean) * inv * g4.z;
    o4.w = (x4.w - mean) * inv * g4.w;
    *(float4*)(out + (size_t)row * DMODEL + tid * 4) = o4;
}

// ---------------------------------------------------------------------------
// Launch: inputs in setup_inputs() order: q, k, v, W_out, g (all f32).
// Scratch layout in d_ws: attn[8192*1024] f32 | proj[8192*1024] f32 (67 MB).
// ---------------------------------------------------------------------------
extern "C" void kernel_launch(void* const* d_in, const int* in_sizes, int n_in,
                              void* d_out, int out_size, void* d_ws, size_t ws_size,
                              hipStream_t stream) {
    (void)in_sizes; (void)n_in; (void)out_size; (void)ws_size;
    const float* q = (const float*)d_in[0];
    const float* k = (const float*)d_in[1];
    const float* v = (const float*)d_in[2];
    const float* W = (const float*)d_in[3];
    const float* g = (const float*)d_in[4];
    float* out  = (float*)d_out;
    float* attn = (float*)d_ws;
    float* proj = attn + (size_t)BATCH * SEQ * DMODEL;

    // 1) attention: (B*H)*(N/64) workgroups, 4 waves each, 24KB LDS
    attn_kernel<<<dim3(BATCH * NHEADS * (SEQ / 64)), dim3(128), 24576, stream>>>(
        q, k, v, attn);
    // 2) projection GEMM: 16 x 128 tiles of 64x64, 8KB LDS
    proj_kernel<<<dim3(DMODEL / 64, (BATCH * SEQ) / 64), dim3(128), 8192, stream>>>(
        attn, W, proj);
    // 3) layernorm: one block per row
    ln_kernel<<<dim3(BATCH * SEQ), dim3(256), 0, stream>>>(proj, g, out);
}
